// QAttNet_62242666053775
// MI455X (gfx1250) — compile-verified, matrix operand order checked by reading, not run on hardware
//
#include <hip/hip_runtime.h>
#include <math.h>

typedef __attribute__((ext_vector_type(16))) _Float16 v16h;
typedef __attribute__((ext_vector_type(8)))  _Float16 v8h;
typedef __attribute__((ext_vector_type(8)))  float    v8f;

#define DIM   100   // embed dim
#define DP    128   // padded K / padded D
#define CTX   128   // context length L
#define ASP   16    // aspect length La
#define BATCH 64
#define HS    136   // f16 LDS row stride in halves (272 B -> rows stay 16B-aligned)
#define US    17    // f32 U-slab row stride (kills 8-way bank conflicts)

// LDS budget (dynamic): 3*128*136 + 2*16*136 halves, 2*128*17 + 2*128 + 8 floats, 144 ints
static constexpr int    kHalfElems = 3 * CTX * HS + 2 * ASP * HS;
static constexpr int    kF32Elems  = 2 * CTX * US + CTX + CTX + 8;
static constexpr int    kIntElems  = CTX + ASP;
static constexpr size_t kSmemBytes = size_t(kHalfElems) * 2 + size_t(kF32Elems) * 4 + size_t(kIntElems) * 4;
static_assert(kSmemBytes < 160 * 1024, "fits 2 workgroups per 320KB WGP");

// CDNA5 16-bit A/B fragment: lane&15 selects the M (or Bt row) index, lane>>4
// selects the K half-pattern {hi*8+0..7, 16+hi*8+0..7}.  Two contiguous
// 8-half (16B) LDS loads per fragment.
__device__ __forceinline__ v16h load_frag(const _Float16* rowptr, int kbase, int hi) {
  const _Float16* p = rowptr + kbase + hi * 8;
  v8h lo = *(const v8h*)(p);
  v8h hb = *(const v8h*)(p + 16);
  return __builtin_shufflevector(lo, hb, 0, 1, 2, 3, 4, 5, 6, 7, 8, 9, 10, 11, 12, 13, 14, 15);
}

__device__ __forceinline__ v8f wmma16(v16h a, v16h b, v8f c) {
  // D = A(16x32 f16) * B(32x16 f16) + C(16x16 f32)
  return __builtin_amdgcn_wmma_f32_16x16x32_f16(false, a, false, b, (short)0, c, false, false);
}

__global__ __launch_bounds__(256, 1)
void qatt_main(const float* __restrict__ amp, const float* __restrict__ phase,
               const float* __restrict__ W, const int* __restrict__ text,
               const int* __restrict__ aspect, float* __restrict__ partials) {
  extern __shared__ char smem_raw[];
  _Float16* sCr = (_Float16*)smem_raw;      // [CTX][HS]  gathered ctx real (f16, K padded)
  _Float16* sCi = sCr + CTX * HS;           // [CTX][HS]  gathered ctx imag
  _Float16* sWs = sCi + CTX * HS;           // [DP][HS]   Wsum rows (== B^T for U GEMM)
  _Float16* sAr = sWs + DP * HS;            // [ASP][HS]  aspect real
  _Float16* sAi = sAr + ASP * HS;           // [ASP][HS]  aspect imag
  float* sUr    = (float*)(sAi + ASP * HS); // [CTX][US]  (Wsum*cr) column slab
  float* sUi    = sUr + CTX * US;           // [CTX][US]  (Wsum*ci) column slab
  float* sS     = sUi + CTX * US;           // [CTX] scores
  float* sAlpha = sS + CTX;                 // [CTX] softmax
  float* sRed   = sAlpha + CTX;             // [8] wave partials
  int*   sIdx   = (int*)(sRed + 8);         // [CTX]
  int*   sAidx  = sIdx + CTX;               // [ASP]

  const int tid  = threadIdx.x;
  const int b    = blockIdx.x;
  const int lane = tid & 31;
  const int wave = tid >> 5;
  const int m16  = lane & 15;
  const int hi   = lane >> 4;

  if (tid < CTX) sIdx[tid] = text[b * CTX + tid];
  if (tid < ASP) sAidx[tid] = aspect[b * ASP + tid];
  __syncthreads();

  // Wsum[d][e] = sum_p W[p][d*DIM+e], zero-padded to 128x128 (f16)
  for (int i = tid; i < DP * DP; i += 256) {
    int d = i >> 7, e = i & 127;
    float v = 0.f;
    if (d < DIM && e < DIM) {
      int o = d * DIM + e;
      v = W[o] + W[DIM * DIM + o] + W[2 * DIM * DIM + o];
    }
    sWs[d * HS + e] = (_Float16)v;
  }
  // gather context complex state: cr = a*cos(p), ci = a*sin(p)
  for (int i = tid; i < CTX * DP; i += 256) {
    int l = i >> 7, dc = i & 127;
    float cr = 0.f, ci = 0.f;
    if (dc < DIM) {
      int idx = sIdx[l];
      float a = amp[idx * DIM + dc];
      float p = phase[idx * DIM + dc];
      cr = a * cosf(p);
      ci = a * sinf(p);
    }
    sCr[l * HS + dc] = (_Float16)cr;
    sCi[l * HS + dc] = (_Float16)ci;
  }
  // gather aspect
  for (int i = tid; i < ASP * DP; i += 256) {
    int m = i >> 7, dc = i & 127;
    float ar = 0.f, ai = 0.f;
    if (dc < DIM) {
      int idx = sAidx[m];
      float a = amp[idx * DIM + dc];
      float p = phase[idx * DIM + dc];
      ar = a * cosf(p);
      ai = a * sinf(p);
    }
    sAr[m * HS + dc] = (_Float16)ar;
    sAi[m * HS + dc] = (_Float16)ai;
  }
  __syncthreads();

  // ---- scores: per wave one 16-row tile; P=Cr*Ar^T Q=Ci*Ai^T R=Cr*Ai^T S=Ci*Ar^T
  {
    v8f P = {}, Q = {}, R = {}, S = {};
    const _Float16* crRow = sCr + (wave * 16 + m16) * HS;
    const _Float16* ciRow = sCi + (wave * 16 + m16) * HS;
    const _Float16* arRow = sAr + m16 * HS;
    const _Float16* aiRow = sAi + m16 * HS;
#pragma unroll
    for (int ks = 0; ks < 4; ++ks) {
      int kb = ks * 32;
      v16h aCr = load_frag(crRow, kb, hi);
      v16h aCi = load_frag(ciRow, kb, hi);
      v16h bAr = load_frag(arRow, kb, hi);
      v16h bAi = load_frag(aiRow, kb, hi);
      P = wmma16(aCr, bAr, P);
      Q = wmma16(aCi, bAi, Q);
      R = wmma16(aCr, bAi, R);
      S = wmma16(aCi, bAr, S);
    }
    // scores[l] = sum_m (P+Q)^2 + (R-S)^2 : horizontal sum over the 16 N-lanes
#pragma unroll
    for (int r = 0; r < 8; ++r) {
      float u = P[r] + Q[r];
      float v = R[r] - S[r];
      float val = u * u + v * v;
      val += __shfl_xor(val, 1, 16);
      val += __shfl_xor(val, 2, 16);
      val += __shfl_xor(val, 4, 16);
      val += __shfl_xor(val, 8, 16);
      if (m16 == 0) sS[wave * 16 + hi * 8 + r] = val;  // lanes 0..15 -> M=r, 16..31 -> M=8+r
    }
  }
  __syncthreads();

  // ---- softmax over L (wave 0)
  if (wave == 0) {
    float mx = -3.4e38f;
    for (int i = lane; i < CTX; i += 32) mx = fmaxf(mx, sS[i]);
    for (int off = 16; off; off >>= 1) mx = fmaxf(mx, __shfl_xor(mx, off, 32));
    float sum = 0.f;
    for (int i = lane; i < CTX; i += 32) {
      float e = expf(sS[i] - mx);
      sAlpha[i] = e;
      sum += e;
    }
    for (int off = 16; off; off >>= 1) sum += __shfl_xor(sum, off, 32);
    float inv = 1.0f / sum;
    for (int i = lane; i < CTX; i += 32) sAlpha[i] *= inv;
  }
  __syncthreads();

  // ---- U = C * Wsum^T (column-tile streamed), fused quadratic form:
  // q_l = (cr+ci)·(Wsum cr) + (ci-cr)·(Wsum ci)
  float qacc = 0.f;
  const _Float16* crRow = sCr + (wave * 16 + m16) * HS;
  const _Float16* ciRow = sCi + (wave * 16 + m16) * HS;
  for (int nt = 0; nt < 8; ++nt) {
    v8f Ur = {}, Ui = {};
    const _Float16* wRow = sWs + (nt * 16 + m16) * HS;  // B^T row d = Wsum row d
#pragma unroll
    for (int ks = 0; ks < 4; ++ks) {
      int kb = ks * 32;
      v16h aCr = load_frag(crRow, kb, hi);
      v16h aCi = load_frag(ciRow, kb, hi);
      v16h bW = load_frag(wRow, kb, hi);
      Ur = wmma16(aCr, bW, Ur);
      Ui = wmma16(aCi, bW, Ui);
    }
#pragma unroll
    for (int r = 0; r < 8; ++r) {   // C/D layout: VGPR r -> M=r (lanes<16) / 8+r
      int m = wave * 16 + hi * 8 + r;
      sUr[m * US + m16] = Ur[r];
      sUi[m * US + m16] = Ui[r];
    }
    __syncthreads();
    if (tid < CTX) {
      const int l = tid;
      float acc = 0.f;
#pragma unroll
      for (int j = 0; j < 16; ++j) {
        float cr = (float)sCr[l * HS + nt * 16 + j];
        float ci = (float)sCi[l * HS + nt * 16 + j];
        acc += (cr + ci) * sUr[l * US + j] + (ci - cr) * sUi[l * US + j];
      }
      qacc += acc;
    }
    __syncthreads();
  }

  // ---- alpha-weighted block reduction (deterministic tree, no float atomics)
  float val = (tid < CTX) ? sAlpha[tid] * qacc : 0.f;
  for (int off = 16; off; off >>= 1) val += __shfl_xor(val, off, 32);
  if (lane == 0) sRed[wave] = val;
  __syncthreads();
  if (tid == 0) {
    float s = 0.f;
#pragma unroll
    for (int i = 0; i < 8; ++i) s += sRed[i];
    partials[b] = s;
  }
}

__global__ void qatt_final(const float* __restrict__ partials, const float* __restrict__ bias,
                           float* __restrict__ out) {
  if (threadIdx.x == 0) {
    float s = 0.f;
    for (int i = 0; i < BATCH; ++i) s += partials[i];  // fixed order -> deterministic
    float bs = bias[0] + bias[1] + bias[2];
    out[0] = s + 2.0f * (float)BATCH * bs;  // the two "+ b" terms summed over (B,P)
  }
}

extern "C" void kernel_launch(void* const* d_in, const int* in_sizes, int n_in,
                              void* d_out, int out_size, void* d_ws, size_t ws_size,
                              hipStream_t stream) {
  (void)in_sizes; (void)n_in; (void)out_size; (void)ws_size;
  const float* amp    = (const float*)d_in[0];  // (V, D) f32
  const float* phase  = (const float*)d_in[1];  // (V, D) f32
  const float* W      = (const float*)d_in[2];  // (P, D*D) f32
  const float* bias   = (const float*)d_in[3];  // (P,) f32
  const int*   text   = (const int*)d_in[4];    // (B, L) i32
  const int*   aspect = (const int*)d_in[5];    // (B, La) i32
  float* out      = (float*)d_out;
  float* partials = (float*)d_ws;               // 64 floats of scratch

  hipFuncSetAttribute((const void*)qatt_main, hipFuncAttributeMaxDynamicSharedMemorySize,
                      (int)kSmemBytes);
  qatt_main<<<BATCH, 256, kSmemBytes, stream>>>(amp, phase, W, text, aspect, partials);
  qatt_final<<<1, 32, 0, stream>>>(partials, bias, out);
}
